// AttentionBlock_60679297958033
// MI455X (gfx1250) — matrix operand order repaired and checked by reading, not compiled, not run on hardware
//
#include <hip/hip_runtime.h>

// ---------------------------------------------------------------------------
// AttentionBlock for MI455X (gfx1250, wave32, WMMA 16x16x32 f16 -> f32 acc)
//   B=8, C=512, H=W=32 (N=1024), heads=8, d=64, groups=8
// Pipeline:
//   1) GroupNorm  -> xnT[b][n][c]            (f16, c contiguous)
//   2) f32->f16 weight conversion
//   3) QKV GEMM   (64x64 tile/wave, 16 WMMA per K-step, base+imm addressing)
//   4) Attention  (32 q-rows/wave, softmax in 64KB LDS, WMMA both matmuls)
//   5) Proj GEMM  (64x64 tile/wave) + bias + residual -> out f32 [b][c][n]
// ---------------------------------------------------------------------------

typedef __attribute__((ext_vector_type(16))) _Float16 h16;
typedef __attribute__((ext_vector_type(8)))  float    f32x8;
typedef __attribute__((ext_vector_type(4)))  float    f32x4;

union FragU { h16 h; f32x4 q[2]; };

static __device__ __forceinline__ f32x8 wmma_f16(h16 a, h16 b, f32x8 c) {
  // D = A(16x32 f16) * B(32x16 f16) + C(16x16 f32)
  return __builtin_amdgcn_wmma_f32_16x16x32_f16(false, a, false, b,
                                                (short)0, c, false, false);
}

#define CB 8
#define CC 512
#define CN 1024
#define CHEADS 8
#define CD 64
#define CG 8

// ---------------------------------------------------------------------------
// 1) GroupNorm over (C/G, H, W) per (b, g); write transposed f16 xnT[b][n][c]
// ---------------------------------------------------------------------------
__global__ void gn_kernel(const float* __restrict__ x,
                          const float* __restrict__ gn_w,
                          const float* __restrict__ gn_b,
                          _Float16* __restrict__ xnT) {
  const int b = blockIdx.x >> 3;
  const int g = blockIdx.x & 7;
  const float* xp = x + (size_t)(b * CC + g * (CC / CG)) * CN;  // 64*1024
  __shared__ float s_sum[256];
  __shared__ float s_sq[256];
  float s = 0.f, sq = 0.f;
  for (int i = threadIdx.x; i < (CC / CG) * CN; i += 256) {
    float v = xp[i];
    s += v; sq += v * v;
  }
  s_sum[threadIdx.x] = s;
  s_sq[threadIdx.x]  = sq;
  __syncthreads();
  for (int off = 128; off > 0; off >>= 1) {
    if ((int)threadIdx.x < off) {
      s_sum[threadIdx.x] += s_sum[threadIdx.x + off];
      s_sq[threadIdx.x]  += s_sq[threadIdx.x + off];
    }
    __syncthreads();
  }
  const float inv_n = 1.0f / (float)((CC / CG) * CN);
  const float mean  = s_sum[0] * inv_n;
  const float var   = s_sq[0] * inv_n - mean * mean;
  const float rstd  = rsqrtf(var + 1e-5f);
  for (int i = threadIdx.x; i < (CC / CG) * CN; i += 256) {
    const int cl = i >> 10;
    const int n  = i & (CN - 1);
    const int c  = g * (CC / CG) + cl;
    const float v = (xp[i] - mean) * rstd * gn_w[c] + gn_b[c];
    xnT[((size_t)(b * CN + n)) * CC + c] = (_Float16)v;
  }
}

// ---------------------------------------------------------------------------
// 2) Convert qkv_w (1536x512) and proj_w (512x512) to f16
// ---------------------------------------------------------------------------
__global__ void cvt_w_kernel(const float* __restrict__ qkv_w,
                             const float* __restrict__ proj_w,
                             _Float16* __restrict__ wq,
                             _Float16* __restrict__ wp) {
  const int i = blockIdx.x * 256 + threadIdx.x;
  if (i < 3 * CC * CC) wq[i] = (_Float16)qkv_w[i];
  if (i < CC * CC)     wp[i] = (_Float16)proj_w[i];
}

// ---------------------------------------------------------------------------
// Shared GEMM core: 64x64 tile per wave, 4x4 register blocking.
// All fragment loads are immediate offsets from TWO base addresses
// (sub-tile row pitch 16*CC and k0 are compile-time after full unroll).
// ---------------------------------------------------------------------------
static __device__ __forceinline__ void gemm_core_64x64(
    const _Float16* __restrict__ abase,   // + (o0+ln)*CC + lh*16
    const _Float16* __restrict__ bbase,   // + (row_n)*CC + lh*16
    f32x8 acc[4][4]) {
#pragma unroll
  for (int sm = 0; sm < 4; ++sm)
#pragma unroll
    for (int sn = 0; sn < 4; ++sn) acc[sm][sn] = (f32x8){};

#pragma unroll
  for (int k0 = 0; k0 < CC; k0 += 32) {
    FragU a[4], bm[4];
#pragma unroll
    for (int i = 0; i < 4; ++i) {
      if ((k0 & 255) == 0) {
        __builtin_prefetch(abase + i * (16 * CC) + k0 + 256, 0, 1);
        __builtin_prefetch(bbase + i * (16 * CC) + k0 + 256, 0, 1);
      }
      a[i].q[0]  = *(const f32x4*)(abase + i * (16 * CC) + k0);
      a[i].q[1]  = *(const f32x4*)(abase + i * (16 * CC) + k0 + 8);
      bm[i].q[0] = *(const f32x4*)(bbase + i * (16 * CC) + k0);
      bm[i].q[1] = *(const f32x4*)(bbase + i * (16 * CC) + k0 + 8);
    }
#pragma unroll
    for (int sm = 0; sm < 4; ++sm)
#pragma unroll
      for (int sn = 0; sn < 4; ++sn)
        acc[sm][sn] = wmma_f16(a[sm].h, bm[sn].h, acc[sm][sn]);
  }
}

// ---------------------------------------------------------------------------
// 3) QKV GEMM: out[o,n] = sum_c W[o,c] * xnT[n,c]; scatter to Q/K/V layouts.
//    Tiles: 8 batches * 24 m-blocks * 16 n-blocks = 3072 waves.
// ---------------------------------------------------------------------------
__global__ __launch_bounds__(128, 1)
void qkv_gemm_kernel(const _Float16* __restrict__ wq,
                     const _Float16* __restrict__ xnT,
                     const float* __restrict__ qkv_b,
                     _Float16* __restrict__ qb,
                     _Float16* __restrict__ kb,
                     _Float16* __restrict__ vT) {
  const int lane = threadIdx.x & 31;
  const int wave = threadIdx.x >> 5;
  int tile = blockIdx.x * 4 + wave;
  const int nt = tile & 15; tile >>= 4;
  const int mt = tile % 24;
  const int b  = tile / 24;
  const int ln = lane & 15;
  const int lh = lane >> 4;
  const int o0 = mt * 64;
  const int n0 = nt * 64;

  const _Float16* abase = wq  + (size_t)(o0 + ln) * CC + lh * 16;
  const _Float16* bbase = xnT + ((size_t)(b * CN + n0 + ln)) * CC + lh * 16;

  f32x8 acc[4][4];
  gemm_core_64x64(abase, bbase, acc);

  // o-block of 64 never crosses a q/k/v (512) or head (64) boundary.
  const int which = o0 / CC;
  const int head  = (o0 % CC) / CD;
#pragma unroll
  for (int sm = 0; sm < 4; ++sm) {
#pragma unroll
    for (int r = 0; r < 8; ++r) {
      const int dd = sm * 16 + r + 8 * lh;      // = o % 64
      const float bias = qkv_b[o0 + dd];
#pragma unroll
      for (int sn = 0; sn < 4; ++sn) {
        const int n = n0 + sn * 16 + ln;
        const _Float16 hv = (_Float16)(acc[sm][sn][r] + bias);
        const size_t qk_idx =
            ((size_t)((b * CHEADS + head) * CN + n)) * CD + dd;
        if (which == 0)      qb[qk_idx] = hv;
        else if (which == 1) kb[qk_idx] = hv;
        else vT[((size_t)((b * CHEADS + head) * CD + dd)) * CN + n] = hv;
      }
    }
  }
}

// ---------------------------------------------------------------------------
// 4) Attention: one wave per (b, h, 32-row q block). S (32x1024 f16) in 64KB
//    LDS. K-frags shared across 2 Q-tiles; V-frags shared across 2 P-tiles.
// ---------------------------------------------------------------------------
__global__ __launch_bounds__(32, 1)
void attn_kernel(const _Float16* __restrict__ qb,
                 const _Float16* __restrict__ kb,
                 const _Float16* __restrict__ vT,
                 _Float16* __restrict__ aoT) {
  __shared__ _Float16 S[32 * CN];   // 64 KB
  const int lane = threadIdx.x & 31;
  const int ln = lane & 15;
  const int lh = lane >> 4;
  int wg = blockIdx.x;              // 8*8*32 = 2048
  const int nt = wg & 31; wg >>= 5;
  const int h  = wg & 7;
  const int b  = wg >> 3;
  const int n0 = nt * 32;
  const size_t bh = (size_t)(b * CHEADS + h);

  // Q A-fragments: two 16-row tiles, d split in two K=32 chunks
  const _Float16* qbase = qb + (bh * CN + n0 + ln) * CD + lh * 16;
  FragU aq[2][2];
#pragma unroll
  for (int t = 0; t < 2; ++t) {
    aq[t][0].q[0] = *(const f32x4*)(qbase + t * (16 * CD));
    aq[t][0].q[1] = *(const f32x4*)(qbase + t * (16 * CD) + 8);
    aq[t][1].q[0] = *(const f32x4*)(qbase + t * (16 * CD) + 32);
    aq[t][1].q[1] = *(const f32x4*)(qbase + t * (16 * CD) + 40);
  }

  // Phase 1: S = (Q K^T) * d^-0.5 ; K-fragments shared by both Q-tiles
  const _Float16* kbase = kb + (bh * CN + ln) * CD + lh * 16;
  for (int mtl = 0; mtl < CN / 16; ++mtl) {
    const _Float16* krow = kbase + mtl * (16 * CD);
    FragU b0, b1;
    b0.q[0] = *(const f32x4*)(krow);
    b0.q[1] = *(const f32x4*)(krow + 8);
    b1.q[0] = *(const f32x4*)(krow + 32);
    b1.q[1] = *(const f32x4*)(krow + 40);
    f32x8 acc0 = (f32x8){}, acc1 = (f32x8){};
    acc0 = wmma_f16(aq[0][0].h, b0.h, acc0);
    acc0 = wmma_f16(aq[0][1].h, b1.h, acc0);
    acc1 = wmma_f16(aq[1][0].h, b0.h, acc1);
    acc1 = wmma_f16(aq[1][1].h, b1.h, acc1);
#pragma unroll
    for (int r = 0; r < 8; ++r) {
      S[(r + 8 * lh) * CN + mtl * 16 + ln]      = (_Float16)(acc0[r] * 0.125f);
      S[(16 + r + 8 * lh) * CN + mtl * 16 + ln] = (_Float16)(acc1[r] * 0.125f);
    }
  }
  __syncthreads();

  // Phase 2: softmax; each lane owns one full row (row == lane)
  float rinv;
  {
    _Float16* sr = S + lane * CN;
    float mx = -3.0e38f;
    for (int i = 0; i < CN; ++i) mx = fmaxf(mx, (float)sr[i]);
    float sum = 0.f;
    for (int i = 0; i < CN; ++i) {
      const float e = __expf((float)sr[i] - mx);
      sum += e;
      sr[i] = (_Float16)e;          // unnormalized prob, in place
    }
    rinv = 1.0f / sum;
  }
  __syncthreads();

  // Phase 3: O = P * V ; V-fragments shared by both P-tiles
  f32x8 oacc[2][4];
#pragma unroll
  for (int t = 0; t < 2; ++t)
#pragma unroll
    for (int dt = 0; dt < 4; ++dt) oacc[t][dt] = (f32x8){};

  const _Float16* vbase = vT + (bh * CD + ln) * CN + lh * 16;
  for (int m0 = 0; m0 < CN; m0 += 32) {
    FragU ap[2];
#pragma unroll
    for (int t = 0; t < 2; ++t) {
      const _Float16* pr = S + (t * 16 + ln) * CN + m0;
      ap[t].q[0] = *(const f32x4*)(pr + lh * 16);       // ds_load_b128
      ap[t].q[1] = *(const f32x4*)(pr + 8 + lh * 16);
    }
#pragma unroll
    for (int dt = 0; dt < 4; ++dt) {
      const _Float16* vr = vbase + dt * (16 * CN) + m0;
      FragU bv;
      bv.q[0] = *(const f32x4*)(vr);
      bv.q[1] = *(const f32x4*)(vr + 8);
      oacc[0][dt] = wmma_f16(ap[0].h, bv.h, oacc[0][dt]);
      oacc[1][dt] = wmma_f16(ap[1].h, bv.h, oacc[1][dt]);
    }
  }

  // Store with 1/rowsum gathered via lane shuffle (lane `row` owns it)
#pragma unroll
  for (int t = 0; t < 2; ++t) {
#pragma unroll
    for (int r = 0; r < 8; ++r) {
      const int row = t * 16 + r + 8 * lh;             // 0..31
      const float ri = __shfl(rinv, row, 32);
      const size_t base = ((size_t)(b * CN + n0 + row)) * CC + h * CD + ln;
#pragma unroll
      for (int dt = 0; dt < 4; ++dt)
        aoT[base + dt * 16] = (_Float16)(oacc[t][dt][r] * ri);
    }
  }
}

// ---------------------------------------------------------------------------
// 5) Proj GEMM + bias + residual; 64x64 tile per wave, 4x4 blocking.
//    Tiles: 8 batches * 8 m-blocks * 16 n-blocks = 1024 waves.
// ---------------------------------------------------------------------------
__global__ __launch_bounds__(128, 1)
void proj_gemm_kernel(const _Float16* __restrict__ wp,
                      const _Float16* __restrict__ aoT,
                      const float* __restrict__ proj_b,
                      const float* __restrict__ x,
                      float* __restrict__ out) {
  const int lane = threadIdx.x & 31;
  const int wave = threadIdx.x >> 5;
  int tile = blockIdx.x * 4 + wave;
  const int nt = tile & 15; tile >>= 4;
  const int mt = tile & 7;
  const int b  = tile >> 3;
  const int ln = lane & 15;
  const int lh = lane >> 4;
  const int o0 = mt * 64;
  const int n0 = nt * 64;

  const _Float16* abase = wp  + (size_t)(o0 + ln) * CC + lh * 16;
  const _Float16* bbase = aoT + ((size_t)(b * CN + n0 + ln)) * CC + lh * 16;

  f32x8 acc[4][4];
  gemm_core_64x64(abase, bbase, acc);

#pragma unroll
  for (int sm = 0; sm < 4; ++sm) {
#pragma unroll
    for (int r = 0; r < 8; ++r) {
      const int o = o0 + sm * 16 + r + 8 * lh;
      const float bias = proj_b[o];
#pragma unroll
      for (int sn = 0; sn < 4; ++sn) {
        const int n = n0 + sn * 16 + ln;
        const size_t idx = ((size_t)(b * CC + o)) * CN + n;
        out[idx] = acc[sm][sn][r] + bias + x[idx];
      }
    }
  }
}

// ---------------------------------------------------------------------------
extern "C" void kernel_launch(void* const* d_in, const int* in_sizes, int n_in,
                              void* d_out, int out_size, void* d_ws, size_t ws_size,
                              hipStream_t stream) {
  (void)in_sizes; (void)n_in; (void)out_size; (void)ws_size;
  const float* x      = (const float*)d_in[0];
  const float* gn_w   = (const float*)d_in[1];
  const float* gn_b   = (const float*)d_in[2];
  const float* qkv_w  = (const float*)d_in[3];
  const float* qkv_b  = (const float*)d_in[4];
  const float* proj_w = (const float*)d_in[5];
  const float* proj_b = (const float*)d_in[6];
  float* out = (float*)d_out;

  // Workspace carve-up (f16 elements): total ~44 MB
  _Float16* ws  = (_Float16*)d_ws;
  _Float16* xnT = ws;                          // 8*1024*512   = 4194304
  _Float16* wq  = xnT + (size_t)CB * CN * CC;  // 1536*512     =  786432
  _Float16* wp  = wq + (size_t)3 * CC * CC;    // 512*512      =  262144
  _Float16* qb  = wp + (size_t)CC * CC;        // 4194304
  _Float16* kb  = qb + (size_t)CB * CC * CN;   // 4194304
  _Float16* vT  = kb + (size_t)CB * CC * CN;   // 4194304
  _Float16* ao  = vT + (size_t)CB * CC * CN;   // 4194304

  gn_kernel<<<CB * CG, 256, 0, stream>>>(x, gn_w, gn_b, xnT);
  cvt_w_kernel<<<(3 * CC * CC) / 256, 256, 0, stream>>>(qkv_w, proj_w, wq, wp);
  // 8 * 24 * 16 = 3072 tiles / 4 waves per block
  qkv_gemm_kernel<<<3072 / 4, 128, 0, stream>>>(wq, xnT, qkv_b, qb, kb, vT);
  // 8 * 8 * 32 row-blocks, one wave each
  attn_kernel<<<CB * CHEADS * (CN / 32), 32, 0, stream>>>(qb, kb, vT, ao);
  // 8 * 8 * 16 = 1024 tiles / 4 waves per block
  proj_gemm_kernel<<<1024 / 4, 128, 0, stream>>>(wp, ao, proj_b, x, out);
}
